// GATv2ClassificationHead_75960791597658
// MI455X (gfx1250) — matrix operand order repaired and checked by reading, not compiled, not run on hardware
//
#include <hip/hip_runtime.h>

#define N_NODES 20000
#define N_EDGES 160000
#define ETOT    (N_EDGES + N_NODES)   // 180000 edges incl. self-loops
#define FEAT    512
#define HEADS   4
#define OUTF    (HEADS * FEAT)        // 2048
#define NTOT    (2 * OUTF)            // 4096 combined columns (W_l | W_r)
#define NEG_SLOPE 0.2f

typedef __attribute__((ext_vector_type(16))) _Float16 v16h;
typedef __attribute__((ext_vector_type(8)))  _Float16 v8h;
typedef __attribute__((ext_vector_type(8)))  float    v8f;

union AFrag { v8h h[2]; v16h v; };
struct Frags { AFrag a; v16h b0, b1, b2, b3; };

// ---------- helpers ----------
__device__ inline void get_edge(const int* __restrict__ ei, int eid, int& src, int& dst) {
    if (eid < N_EDGES) { src = ei[eid]; dst = ei[N_EDGES + eid]; }
    else               { src = dst = eid - N_EDGES; }     // self-loop
}
// order-preserving float->uint encoding for atomicMax
__device__ inline unsigned fenc(float f) {
    unsigned u = __float_as_uint(f);
    return (u & 0x80000000u) ? ~u : (u | 0x80000000u);
}
__device__ inline float fdec(unsigned e) {
    unsigned u = (e & 0x80000000u) ? (e & 0x7FFFFFFFu) : ~e;
    return __uint_as_float(u);
}

// ---------- precision conversion ----------
__global__ void __launch_bounds__(256) gat_conv_x(const float* __restrict__ x, _Float16* __restrict__ xb) {
    int i = blockIdx.x * 256 + threadIdx.x;              // exactly N_NODES*FEAT threads
    xb[i] = (_Float16)x[i];
}
// wt[(which*2048 + n)*512 + k] = W[k*2048 + n]   (pre-transposed so B frags are contiguous)
__global__ void __launch_bounds__(256) gat_conv_w(const float* __restrict__ Wl,
                                                  const float* __restrict__ Wr,
                                                  _Float16* __restrict__ wt) {
    int idx = blockIdx.x * 256 + threadIdx.x;            // 2 * 2048 * 512 = 2^21 threads
    int which = idx >> 20;                               // 2048*512 = 2^20
    int rem   = idx & 0xFFFFF;
    int n = rem >> 9;
    int k = rem & 511;
    const float* W = which ? Wr : Wl;
    wt[idx] = (_Float16)W[(size_t)k * OUTF + n];
}

// ---------- WMMA GEMM: [20000 x 512] (f16) x [512 x 4096] (f16) -> xl|xr (f32) ----------
__device__ inline void load_frags(Frags& f, const _Float16* __restrict__ arow,
                                  const _Float16* __restrict__ b0, const _Float16* __restrict__ b1,
                                  const _Float16* __restrict__ b2, const _Float16* __restrict__ b3,
                                  int k0, int kh) {
    f.a.h[0] = *(const v8h*)(arow + k0 + kh * 16);       // A 16x32: contiguous 16 elems per lane
    f.a.h[1] = *(const v8h*)(arow + k0 + kh * 16 + 8);
    f.b0 = *(const v16h*)(b0 + k0);                       // B 32x16: contiguous 32B per lane
    f.b1 = *(const v16h*)(b1 + k0);
    f.b2 = *(const v16h*)(b2 + k0);
    f.b3 = *(const v16h*)(b3 + k0);
}

__global__ void __launch_bounds__(256)
gat_gemm_wmma(const _Float16* __restrict__ xb, const _Float16* __restrict__ wt,
              float* __restrict__ xl, float* __restrict__ xr) {
    const int wave = (blockIdx.x << 3) + (threadIdx.x >> 5);   // 80000 waves total
    const int lane = threadIdx.x & 31;
    const int mt = wave >> 6;        // 0..1249  (M tile of 16 rows)
    const int ng = wave & 63;        // 0..63    (group of 64 columns)
    const int mi = lane & 15;
    const int kh = lane >> 4;

    const _Float16* arow = xb + (size_t)(mt * 16 + mi) * FEAT;
    const int ncol0 = ng * 64 + mi;
    const _Float16* b0 = wt + (size_t)(ncol0     ) * FEAT + kh * 16;
    const _Float16* b1 = wt + (size_t)(ncol0 + 16) * FEAT + kh * 16;
    const _Float16* b2 = wt + (size_t)(ncol0 + 32) * FEAT + kh * 16;
    const _Float16* b3 = wt + (size_t)(ncol0 + 48) * FEAT + kh * 16;

    v8f acc0 = {}, acc1 = {}, acc2 = {}, acc3 = {};

    // one-deep software pipeline: prefetch k+32 fragments before consuming k,
    // so the 10 outstanding loads overlap the 4 WMMAs instead of a full drain.
    Frags cur, nxt;
    load_frags(cur, arow, b0, b1, b2, b3, 0, kh);
    #pragma unroll 3
    for (int k0 = 32; k0 < FEAT; k0 += 32) {
        load_frags(nxt, arow, b0, b1, b2, b3, k0, kh);
        acc0 = __builtin_amdgcn_wmma_f32_16x16x32_f16(false, cur.a.v, false, cur.b0, (short)0, acc0, false, false);
        acc1 = __builtin_amdgcn_wmma_f32_16x16x32_f16(false, cur.a.v, false, cur.b1, (short)0, acc1, false, false);
        acc2 = __builtin_amdgcn_wmma_f32_16x16x32_f16(false, cur.a.v, false, cur.b2, (short)0, acc2, false, false);
        acc3 = __builtin_amdgcn_wmma_f32_16x16x32_f16(false, cur.a.v, false, cur.b3, (short)0, acc3, false, false);
        cur = nxt;
    }
    acc0 = __builtin_amdgcn_wmma_f32_16x16x32_f16(false, cur.a.v, false, cur.b0, (short)0, acc0, false, false);
    acc1 = __builtin_amdgcn_wmma_f32_16x16x32_f16(false, cur.a.v, false, cur.b1, (short)0, acc1, false, false);
    acc2 = __builtin_amdgcn_wmma_f32_16x16x32_f16(false, cur.a.v, false, cur.b2, (short)0, acc2, false, false);
    acc3 = __builtin_amdgcn_wmma_f32_16x16x32_f16(false, cur.a.v, false, cur.b3, (short)0, acc3, false, false);

    // C layout: VGPR r -> M = r + 8*kh, N = mi
    const int colbase = ng * 64;
    float* outp; int cofs;
    if (colbase < OUTF) { outp = xl; cofs = colbase; }
    else                { outp = xr; cofs = colbase - OUTF; }
    #pragma unroll
    for (int r = 0; r < 8; ++r) {
        const size_t row = (size_t)(mt * 16 + r + 8 * kh) * OUTF;
        outp[row + cofs      + mi] = acc0[r];
        outp[row + cofs + 16 + mi] = acc1[r];
        outp[row + cofs + 32 + mi] = acc2[r];
        outp[row + cofs + 48 + mi] = acc3[r];
    }
}

// ---------- edge scores: e = att . leaky_relu(xl[src] + xr[dst]) ----------
__global__ void __launch_bounds__(256)
gat_scores(const int* __restrict__ ei, const float* __restrict__ xl, const float* __restrict__ xr,
           const float* __restrict__ att, float* __restrict__ scores) {
    const int task = blockIdx.x * 8 + (threadIdx.x >> 5);      // (edge, head) per wave
    const int lane = threadIdx.x & 31;
    const int eid = task >> 2, h = task & 3;
    int src, dst; get_edge(ei, eid, src, dst);

    const float* pl = xl + (size_t)src * OUTF + h * FEAT + lane * 16;
    const float* pr = xr + (size_t)dst * OUTF + h * FEAT + lane * 16;
    const float* pa = att + h * FEAT + lane * 16;
    float s = 0.f;
    #pragma unroll
    for (int t = 0; t < 4; ++t) {
        float4 a = *(const float4*)(pl + t * 4);
        float4 b = *(const float4*)(pr + t * 4);
        float4 w = *(const float4*)(pa + t * 4);
        float u;
        u = a.x + b.x; u = u > 0.f ? u : NEG_SLOPE * u; s += u * w.x;
        u = a.y + b.y; u = u > 0.f ? u : NEG_SLOPE * u; s += u * w.y;
        u = a.z + b.z; u = u > 0.f ? u : NEG_SLOPE * u; s += u * w.z;
        u = a.w + b.w; u = u > 0.f ? u : NEG_SLOPE * u; s += u * w.w;
    }
    #pragma unroll
    for (int off = 16; off; off >>= 1) s += __shfl_xor(s, off, 32);
    if (lane == 0) scores[task] = s;
}

// ---------- segment softmax ----------
__global__ void __launch_bounds__(256) gat_init_nh(unsigned* __restrict__ maxenc, float* __restrict__ denom) {
    int i = blockIdx.x * 256 + threadIdx.x;
    if (i < N_NODES * HEADS) { maxenc[i] = 0x007FFFFFu; /* enc(-inf) */ denom[i] = 0.f; }
}
__global__ void __launch_bounds__(256)
gat_segmax(const int* __restrict__ ei, const float* __restrict__ scores, unsigned* __restrict__ maxenc) {
    int t = blockIdx.x * 256 + threadIdx.x;
    if (t >= ETOT * HEADS) return;
    int eid = t >> 2, h = t & 3;
    int src, dst; get_edge(ei, eid, src, dst);
    atomicMax(&maxenc[dst * HEADS + h], fenc(scores[t]));
}
__global__ void __launch_bounds__(256)
gat_alpha(const int* __restrict__ ei, float* __restrict__ scores,
          const unsigned* __restrict__ maxenc, float* __restrict__ denom) {
    int t = blockIdx.x * 256 + threadIdx.x;
    if (t >= ETOT * HEADS) return;
    int eid = t >> 2, h = t & 3;
    int src, dst; get_edge(ei, eid, src, dst);
    float a = __expf(scores[t] - fdec(maxenc[dst * HEADS + h]));
    scores[t] = a;                                  // alpha stored in place
    atomicAdd(&denom[dst * HEADS + h], a);
}

// ---------- aggregation ----------
__global__ void __launch_bounds__(256) gat_zero(float* __restrict__ p) {
    p[(size_t)blockIdx.x * 256 + threadIdx.x] = 0.f;     // exactly N_NODES*OUTF threads
}
__global__ void __launch_bounds__(256)
gat_aggregate(const int* __restrict__ ei, const float* __restrict__ xl,
              const float* __restrict__ alpha, const float* __restrict__ denom,
              float* __restrict__ agg) {
    const int task = blockIdx.x * 8 + (threadIdx.x >> 5);
    const int lane = threadIdx.x & 31;
    const int eid = task >> 2, h = task & 3;
    int src, dst; get_edge(ei, eid, src, dst);
    const float coef = alpha[task] / denom[dst * HEADS + h];
    const float* xsrc = xl + (size_t)src * OUTF + h * FEAT + lane * 16;
    float*       od   = agg + (size_t)dst * OUTF + h * FEAT + lane * 16;
    #pragma unroll
    for (int t = 0; t < 4; ++t) {
        float4 v = *(const float4*)(xsrc + t * 4);        // b128 gather
        atomicAdd(&od[t * 4 + 0], coef * v.x);
        atomicAdd(&od[t * 4 + 1], coef * v.y);
        atomicAdd(&od[t * 4 + 2], coef * v.z);
        atomicAdd(&od[t * 4 + 3], coef * v.w);
    }
}

// ---------- head mean + bias + classifier ----------
__global__ void __launch_bounds__(256)
gat_head(const float* __restrict__ agg, const float* __restrict__ bias,
         const float* __restrict__ Wc, const float* __restrict__ bc, float* __restrict__ out) {
    const int i = blockIdx.x * 8 + (threadIdx.x >> 5);
    const int lane = threadIdx.x & 31;
    const float* base = agg + (size_t)i * OUTF + lane * 16;
    float s0 = 0.f, s1 = 0.f;
    #pragma unroll
    for (int t = 0; t < 16; ++t) {
        int c = lane * 16 + t;
        float v = 0.25f * (base[t] + base[512 + t] + base[1024 + t] + base[1536 + t]) + bias[c];
        s0 += v * Wc[c * 2 + 0];
        s1 += v * Wc[c * 2 + 1];
    }
    #pragma unroll
    for (int off = 16; off; off >>= 1) {
        s0 += __shfl_xor(s0, off, 32);
        s1 += __shfl_xor(s1, off, 32);
    }
    if (lane == 0) { out[i * 2 + 0] = s0 + bc[0]; out[i * 2 + 1] = s1 + bc[1]; }
}

// ---------- launch ----------
extern "C" void kernel_launch(void* const* d_in, const int* in_sizes, int n_in,
                              void* d_out, int out_size, void* d_ws, size_t ws_size,
                              hipStream_t stream) {
    const float* x    = (const float*)d_in[0];
    const int*   ei   = (const int*)  d_in[1];
    const float* Wl   = (const float*)d_in[2];
    const float* Wr   = (const float*)d_in[3];
    const float* att  = (const float*)d_in[4];
    const float* bias = (const float*)d_in[5];
    const float* Wc   = (const float*)d_in[6];
    const float* bc   = (const float*)d_in[7];
    float* out = (float*)d_out;

    char* ws = (char*)d_ws;
    size_t off = 0;
    auto carve = [&](size_t bytes) { void* p = ws + off; off = (off + bytes + 255) & ~(size_t)255; return p; };
    _Float16* xb     = (_Float16*)carve((size_t)N_NODES * FEAT * 2);   // x in f16
    _Float16* wt     = (_Float16*)carve((size_t)NTOT * FEAT * 2);      // [W_l|W_r]^T in f16
    float*    xl     = (float*)   carve((size_t)N_NODES * OUTF * 4);
    float*    xr     = (float*)   carve((size_t)N_NODES * OUTF * 4);   // reused as agg after scores
    float*    scores = (float*)   carve((size_t)ETOT * HEADS * 4);     // alpha stored in place
    unsigned* maxenc = (unsigned*)carve((size_t)N_NODES * HEADS * 4);
    float*    denom  = (float*)   carve((size_t)N_NODES * HEADS * 4);
    float*    agg    = xr;                                             // alias: xr dead after gat_scores

    gat_conv_x   <<<(N_NODES * FEAT) / 256, 256, 0, stream>>>(x, xb);
    gat_conv_w   <<<(2 * 2048 * 512) / 256, 256, 0, stream>>>(Wl, Wr, wt);
    gat_gemm_wmma<<<(N_NODES / 16) * (NTOT / 64) / 8, 256, 0, stream>>>(xb, wt, xl, xr);
    gat_scores   <<<(ETOT * HEADS) / 8 / 32, 256, 0, stream>>>(ei, xl, xr, att, scores);
    gat_init_nh  <<<(N_NODES * HEADS + 255) / 256, 256, 0, stream>>>(maxenc, denom);
    gat_segmax   <<<(ETOT * HEADS + 255) / 256, 256, 0, stream>>>(ei, scores, maxenc);
    gat_alpha    <<<(ETOT * HEADS + 255) / 256, 256, 0, stream>>>(ei, scores, maxenc, denom);
    gat_zero     <<<(N_NODES * OUTF) / 256, 256, 0, stream>>>(agg);
    gat_aggregate<<<(ETOT * HEADS) / 8 / 32, 256, 0, stream>>>(ei, xl, scores, denom, agg);
    gat_head     <<<N_NODES / 8, 256, 0, stream>>>(agg, bias, Wc, bc, out);
}
